// rnn_forward_model_v3_44495861187269
// MI455X (gfx1250) — compile-verified, matrix operand order checked by reading, not run on hardware
//
#include <hip/hip_runtime.h>
#include <hip/hip_bf16.h>

typedef __attribute__((ext_vector_type(16))) _Float16 v16h;
typedef __attribute__((ext_vector_type(8)))  _Float16 v8h;
typedef __attribute__((ext_vector_type(8)))  float    v8f;
typedef __attribute__((ext_vector_type(4)))  float    v4f;
typedef _Float16 h16;

// k-index within a 32-wide K tile for A/B fragment element e (0..15), lane-group g (0/1)
#define KIDX(e,g) ((((e) >> 3) * 16) + ((g) * 8) + ((e) & 7))

// Fragment-packed weight sizes (halfs): [KT * NT * 32 lanes * 16 elems]
#define PK1A (1 * 4 * 512)
#define PK1B (2 * 2 * 512)
#define PK2A (2 * 12 * 512)
#define PK2B (2 * 12 * 512)
#define PK3A (6 * 4 * 512)
#define PK3B (2 * 2 * 512)
#define PK4A (4 * 4 * 512)
#define PK4B (2 * 1 * 512)

__device__ __forceinline__ v8f wmma16(v16h a, v16h b, v8f c) {
  return __builtin_amdgcn_wmma_f32_16x16x32_f16(false, a, false, b, (short)0, c, false, false);
}

__device__ __forceinline__ v16h cat8(v8h lo, v8h hi) {
  return __builtin_shufflevector(lo, hi, 0,1,2,3,4,5,6,7,8,9,10,11,12,13,14,15);
}

// ---- CDNA5 async global->LDS copy (ASYNCcnt-tracked, bypasses VGPRs) ----
__device__ __forceinline__ void async_copy_b128(h16* ldsDst, const h16* gsrc) {
  // Flat address of an LDS location carries the LDS byte offset in addr[31:0]
  unsigned lds_off = (unsigned)(uintptr_t)ldsDst;
  asm volatile("global_load_async_to_lds_b128 %0, %1, off"
               :: "v"(lds_off), "v"(gsrc) : "memory");
}
__device__ __forceinline__ void stage_async(h16* lds, const h16* __restrict__ g, int halfs) {
  for (int i = threadIdx.x * 8; i < halfs; i += blockDim.x * 8)
    async_copy_b128(lds + i, g + i);
}
__device__ __forceinline__ void async_wait_barrier() {
  asm volatile("s_wait_asynccnt 0" ::: "memory");
  __syncthreads();
}

// Packed B-fragment load: one contiguous 32B read per lane (2 x ds_load_b128)
__device__ __forceinline__ v16h load_bp(const h16* lds, int frag, int lane) {
  return *(const v16h*)(lds + (size_t)(frag * 32 + lane) * 16);
}

// A-fragment from a row-major [16 x Kcols] f16 LDS activation tile: two b128 reads
__device__ __forceinline__ v16h load_a_row(const h16* ldsH, int kt, int Kcols, int lane) {
  int g = lane >> 4, m = lane & 15;
  const h16* p = ldsH + m * Kcols + kt * 32 + g * 8;
  v8h lo = *(const v8h*)(p);
  v8h hi = *(const v8h*)(p + 16);
  return cat8(lo, hi);
}

__device__ __forceinline__ float sigmoidf_(float v) { return 1.f / (1.f + __expf(-v)); }

// ---------------- Prepack: fp32 W[n_out][k_in] -> fragment-packed f16 in global ws ------------
__global__ __launch_bounds__(256)
void prepack_kernel(const float* __restrict__ W, h16* __restrict__ dst,
                    int n_out, int k_in, int KT, int NT) {
  int total = KT * NT * 512;
  for (int i = blockIdx.x * blockDim.x + threadIdx.x; i < total; i += gridDim.x * blockDim.x) {
    int e    = i & 15;
    int lane = (i >> 4) & 31;
    int frag = i >> 9;            // kt*NT + nt
    int nt = frag % NT, kt = frag / NT;
    int g = lane >> 4;
    int k = kt * 32 + KIDX(e, g);
    int n = nt * 16 + (lane & 15);
    h16 v = (h16)0.f;
    if (k < k_in && n < n_out) v = (h16)W[n * k_in + k];
    dst[i] = v;
  }
}

// ---------------- Kernel 1: GN1 edge MLP (20 -> 64 -> 32) + scatter-add agg1 ----------------
// Also emits padded per-edge feature block ef[E][48] = [ea(4) | 0(4) | e1(32) | 0(8)] (f16)
__global__ __launch_bounds__(256)
void gn1_edge_kernel(const float* __restrict__ x, const int* __restrict__ ei,
                     const float* __restrict__ ea,
                     const h16* __restrict__ pkW1, const float* __restrict__ b1,
                     const h16* __restrict__ pkW2, const float* __restrict__ b2,
                     h16* __restrict__ ef, float* __restrict__ agg1,
                     int N, int E) {
  __shared__ alignas(32) h16 ldsW1[PK1A];
  __shared__ alignas(32) h16 ldsW2[PK1B];
  __shared__ alignas(32) h16 ldsH[8][16 * 64];
  __shared__ int ldsD[8][16];
  stage_async(ldsW1, pkW1, PK1A);
  stage_async(ldsW2, pkW2, PK1B);
  async_wait_barrier();

  int wave = threadIdx.x >> 5, lane = threadIdx.x & 31;
  int tile = blockIdx.x * (blockDim.x >> 5) + wave;
  int eBase = tile * 16;
  if (eBase >= E) return;
  int g = lane >> 4, m = lane & 15;
  int eIdx = min(eBase + m, E - 1);
  int src = ei[eIdx];
  int dst = ei[E + eIdx];
  if (g == 0) ldsD[wave][m] = dst;

  // A fragment (K=20 pad 32): g0: k0..7 = x[src], k16..23 = ea+pad; g1: k8..15 = x[dst]
  v8h lo, hi;
  if (g == 0) {
    lo = __builtin_convertvector(*(const v8f*)(x + (size_t)src * 8), v8h);
    v4f e4 = *(const v4f*)(ea + (size_t)eIdx * 4);
    v8h t = {(h16)e4.x, (h16)e4.y, (h16)e4.z, (h16)e4.w,
             (h16)0.f, (h16)0.f, (h16)0.f, (h16)0.f};
    hi = t;
    *(v8h*)(ef + (size_t)eIdx * 48) = t;          // ef[0..7] = ea + pad
  } else {
    lo = __builtin_convertvector(*(const v8f*)(x + (size_t)dst * 8), v8h);
    v8h z = {};
    hi = z;
    *(v8h*)(ef + (size_t)eIdx * 48 + 40) = z;     // ef[40..47] = pad
  }
  v16h a = cat8(lo, hi);

  #pragma unroll
  for (int nt = 0; nt < 4; ++nt) {
    v8f c = {};
    c = wmma16(a, load_bp(ldsW1, nt, lane), c);
    float bias = b1[nt * 16 + m];
    #pragma unroll
    for (int j = 0; j < 8; ++j) {
      int row = j + 8 * g;
      ldsH[wave][row * 64 + nt * 16 + m] = (h16)fmaxf(c[j] + bias, 0.f);
    }
  }

  v8f c2[2] = {{}, {}};
  #pragma unroll
  for (int kt = 0; kt < 2; ++kt) {
    v16h a2 = load_a_row(ldsH[wave], kt, 64, lane);
    #pragma unroll
    for (int nt = 0; nt < 2; ++nt)
      c2[nt] = wmma16(a2, load_bp(ldsW2, kt * 2 + nt, lane), c2[nt]);
  }
  #pragma unroll
  for (int nt = 0; nt < 2; ++nt) {
    float bias = b2[nt * 16 + m];
    #pragma unroll
    for (int j = 0; j < 8; ++j) {
      int row = j + 8 * g;
      int edge = eBase + row;
      if (edge < E) {
        float v = c2[nt][j] + bias;
        ef[(size_t)edge * 48 + 8 + nt * 16 + m] = (h16)v;
        atomicAdd(&agg1[(size_t)ldsD[wave][row] * 32 + nt * 16 + m], v);
      }
    }
  }
}

// ---------------- Kernel 2: GRU cell per node + build x1c (f16) ----------------
template<int K0>
__device__ __forceinline__ v8h run_in1(const float* __restrict__ x,
                                       const float* __restrict__ agg1, int node) {
  if constexpr (K0 < 8)
    return __builtin_convertvector(*(const v8f*)(x + (size_t)node * 8), v8h);
  else if constexpr (K0 < 40)
    return __builtin_convertvector(*(const v8f*)(agg1 + (size_t)node * 32 + (K0 - 8)), v8h);
  else { v8h z = {}; return z; }
}
template<int K0>
__device__ __forceinline__ v8h run_h(const float* __restrict__ hidden, int node) {
  return __builtin_convertvector(*(const v8f*)(hidden + (size_t)node * 64 + K0), v8h);
}

__global__ __launch_bounds__(256)
void gru_node_kernel(const float* __restrict__ x, const float* __restrict__ hidden,
                     const float* __restrict__ agg1,
                     const h16* __restrict__ pkWih, const float* __restrict__ bih,
                     const h16* __restrict__ pkWhh, const float* __restrict__ bhh,
                     float* __restrict__ hOut, h16* __restrict__ x1c, int N) {
  __shared__ alignas(32) h16 ldsWih[PK2A];
  __shared__ alignas(32) h16 ldsWhh[PK2B];
  stage_async(ldsWih, pkWih, PK2A);
  stage_async(ldsWhh, pkWhh, PK2B);
  async_wait_barrier();

  int wave = threadIdx.x >> 5, lane = threadIdx.x & 31;
  int tile = blockIdx.x * (blockDim.x >> 5) + wave;
  int nBase = tile * 16;
  if (nBase >= N) return;
  int g = lane >> 4, m = lane & 15;
  int node = min(nBase + m, N - 1);

  if (lane < 16 && nBase + lane < N) {
    int nd = nBase + lane;
    *(v8h*)(x1c + (size_t)nd * 72) =
        __builtin_convertvector(*(const v8f*)(x + (size_t)nd * 8), v8h);
  }

  v16h a1[2], ah[2];
  if (g == 0) {
    a1[0] = cat8(run_in1<0>(x, agg1, node), run_in1<16>(x, agg1, node));
    a1[1] = cat8(run_in1<32>(x, agg1, node), run_in1<48>(x, agg1, node));
    ah[0] = cat8(run_h<0 >(hidden, node), run_h<16>(hidden, node));
    ah[1] = cat8(run_h<32>(hidden, node), run_h<48>(hidden, node));
  } else {
    a1[0] = cat8(run_in1<8 >(x, agg1, node), run_in1<24>(x, agg1, node));
    a1[1] = cat8(run_in1<40>(x, agg1, node), run_in1<56>(x, agg1, node));
    ah[0] = cat8(run_h<8 >(hidden, node), run_h<24>(hidden, node));
    ah[1] = cat8(run_h<40>(hidden, node), run_h<56>(hidden, node));
  }

  #pragma unroll
  for (int nt = 0; nt < 4; ++nt) {
    v8f cir = {}, ciz = {}, cin = {}, chr_ = {}, chz = {}, chn = {};
    #pragma unroll
    for (int kt = 0; kt < 2; ++kt) {
      cir  = wmma16(a1[kt], load_bp(ldsWih, kt * 12 + nt,     lane), cir);
      ciz  = wmma16(a1[kt], load_bp(ldsWih, kt * 12 + nt + 4, lane), ciz);
      cin  = wmma16(a1[kt], load_bp(ldsWih, kt * 12 + nt + 8, lane), cin);
      chr_ = wmma16(ah[kt], load_bp(ldsWhh, kt * 12 + nt,     lane), chr_);
      chz  = wmma16(ah[kt], load_bp(ldsWhh, kt * 12 + nt + 4, lane), chz);
      chn  = wmma16(ah[kt], load_bp(ldsWhh, kt * 12 + nt + 8, lane), chn);
    }
    int u = nt * 16 + m;
    float bir = bih[u], biz = bih[64 + u], bin = bih[128 + u];
    float bhr = bhh[u], bhz = bhh[64 + u], bhn = bhh[128 + u];
    #pragma unroll
    for (int j = 0; j < 8; ++j) {
      int row = j + 8 * g;
      int nd = nBase + row;
      if (nd < N) {
        float r  = sigmoidf_(cir[j] + bir + chr_[j] + bhr);
        float z  = sigmoidf_(ciz[j] + biz + chz[j] + bhz);
        float hn = tanhf(cin[j] + bin + r * (chn[j] + bhn));
        float ho = hidden[(size_t)nd * 64 + u];
        float hN = (1.f - z) * hn + z * ho;
        hOut[(size_t)nd * 64 + u] = hN;
        x1c[(size_t)nd * 72 + 8 + u] = (h16)hN;
      }
    }
  }
}

// ---------------- Kernel 3: GN2 edge MLP (180 pad 192 -> 64 -> 32) + scatter-add agg2 -------
template<int K0>
__device__ __forceinline__ v8h run3(const h16* __restrict__ x1c, const h16* __restrict__ ef,
                                    int src, int dst, int edge) {
  if constexpr (K0 < 72)
    return *(const v8h*)(x1c + (size_t)src * 72 + K0);
  else if constexpr (K0 < 144)
    return *(const v8h*)(x1c + (size_t)dst * 72 + (K0 - 72));
  else
    return *(const v8h*)(ef + (size_t)edge * 48 + (K0 - 144));
}

__global__ __launch_bounds__(256)
void gn2_edge_kernel(const int* __restrict__ ei,
                     const h16* __restrict__ x1c, const h16* __restrict__ ef,
                     const h16* __restrict__ pkW1, const float* __restrict__ b1,
                     const h16* __restrict__ pkW2, const float* __restrict__ b2,
                     float* __restrict__ agg2, int N, int E) {
  __shared__ alignas(32) h16 ldsW1[PK3A];
  __shared__ alignas(32) h16 ldsW2[PK3B];
  __shared__ alignas(32) h16 ldsH[8][16 * 64];
  __shared__ int ldsD[8][16];
  stage_async(ldsW1, pkW1, PK3A);
  stage_async(ldsW2, pkW2, PK3B);
  async_wait_barrier();

  int wave = threadIdx.x >> 5, lane = threadIdx.x & 31;
  int tile = blockIdx.x * (blockDim.x >> 5) + wave;
  int eBase = tile * 16;
  if (eBase >= E) return;
  int g = lane >> 4, m = lane & 15;
  int eIdx = min(eBase + m, E - 1);
  int src = ei[eIdx], dst = ei[E + eIdx];
  if (g == 0) ldsD[wave][m] = dst;

  v8f c1[4] = {{}, {}, {}, {}};
#define GN2_STEP(KT) { \
    v16h a; \
    if (g == 0) a = cat8(run3<(KT)*32+0>(x1c, ef, src, dst, eIdx), \
                         run3<(KT)*32+16>(x1c, ef, src, dst, eIdx)); \
    else        a = cat8(run3<(KT)*32+8>(x1c, ef, src, dst, eIdx), \
                         run3<(KT)*32+24>(x1c, ef, src, dst, eIdx)); \
    c1[0] = wmma16(a, load_bp(ldsW1, (KT)*4 + 0, lane), c1[0]); \
    c1[1] = wmma16(a, load_bp(ldsW1, (KT)*4 + 1, lane), c1[1]); \
    c1[2] = wmma16(a, load_bp(ldsW1, (KT)*4 + 2, lane), c1[2]); \
    c1[3] = wmma16(a, load_bp(ldsW1, (KT)*4 + 3, lane), c1[3]); }
  GN2_STEP(0) GN2_STEP(1) GN2_STEP(2) GN2_STEP(3) GN2_STEP(4) GN2_STEP(5)
#undef GN2_STEP

  #pragma unroll
  for (int nt = 0; nt < 4; ++nt) {
    float bias = b1[nt * 16 + m];
    #pragma unroll
    for (int j = 0; j < 8; ++j) {
      int row = j + 8 * g;
      ldsH[wave][row * 64 + nt * 16 + m] = (h16)fmaxf(c1[nt][j] + bias, 0.f);
    }
  }

  v8f c2[2] = {{}, {}};
  #pragma unroll
  for (int kt = 0; kt < 2; ++kt) {
    v16h a2 = load_a_row(ldsH[wave], kt, 64, lane);
    #pragma unroll
    for (int nt = 0; nt < 2; ++nt)
      c2[nt] = wmma16(a2, load_bp(ldsW2, kt * 2 + nt, lane), c2[nt]);
  }
  #pragma unroll
  for (int nt = 0; nt < 2; ++nt) {
    float bias = b2[nt * 16 + m];
    #pragma unroll
    for (int j = 0; j < 8; ++j) {
      int row = j + 8 * g;
      int edge = eBase + row;
      if (edge < E) {
        float v = c2[nt][j] + bias;
        atomicAdd(&agg2[(size_t)ldsD[wave][row] * 32 + nt * 16 + m], v);
      }
    }
  }
}

// ---------------- Kernel 4: output MLP (104 pad 128 -> 64 -> 7) ----------------
template<int K0>
__device__ __forceinline__ v8h run4(const h16* __restrict__ x1c,
                                    const float* __restrict__ agg2, int node) {
  if constexpr (K0 < 72)
    return *(const v8h*)(x1c + (size_t)node * 72 + K0);
  else if constexpr (K0 < 104)
    return __builtin_convertvector(*(const v8f*)(agg2 + (size_t)node * 32 + (K0 - 72)), v8h);
  else { v8h z = {}; return z; }
}

__global__ __launch_bounds__(256)
void out_node_kernel(const h16* __restrict__ x1c, const float* __restrict__ agg2,
                     const h16* __restrict__ pkW1, const float* __restrict__ b1,
                     const h16* __restrict__ pkW2, const float* __restrict__ b2,
                     float* __restrict__ out, int N) {
  __shared__ alignas(32) h16 ldsW1[PK4A];
  __shared__ alignas(32) h16 ldsW2[PK4B];
  __shared__ alignas(32) h16 ldsH[8][16 * 64];
  stage_async(ldsW1, pkW1, PK4A);
  stage_async(ldsW2, pkW2, PK4B);
  async_wait_barrier();

  int wave = threadIdx.x >> 5, lane = threadIdx.x & 31;
  int tile = blockIdx.x * (blockDim.x >> 5) + wave;
  int nBase = tile * 16;
  if (nBase >= N) return;
  int g = lane >> 4, m = lane & 15;
  int node = min(nBase + m, N - 1);

  v8f c1[4] = {{}, {}, {}, {}};
#define OUT_STEP(KT) { \
    v16h a; \
    if (g == 0) a = cat8(run4<(KT)*32+0>(x1c, agg2, node), \
                         run4<(KT)*32+16>(x1c, agg2, node)); \
    else        a = cat8(run4<(KT)*32+8>(x1c, agg2, node), \
                         run4<(KT)*32+24>(x1c, agg2, node)); \
    c1[0] = wmma16(a, load_bp(ldsW1, (KT)*4 + 0, lane), c1[0]); \
    c1[1] = wmma16(a, load_bp(ldsW1, (KT)*4 + 1, lane), c1[1]); \
    c1[2] = wmma16(a, load_bp(ldsW1, (KT)*4 + 2, lane), c1[2]); \
    c1[3] = wmma16(a, load_bp(ldsW1, (KT)*4 + 3, lane), c1[3]); }
  OUT_STEP(0) OUT_STEP(1) OUT_STEP(2) OUT_STEP(3)
#undef OUT_STEP

  #pragma unroll
  for (int nt = 0; nt < 4; ++nt) {
    float bias = b1[nt * 16 + m];
    #pragma unroll
    for (int j = 0; j < 8; ++j) {
      int row = j + 8 * g;
      ldsH[wave][row * 64 + nt * 16 + m] = (h16)fmaxf(c1[nt][j] + bias, 0.f);
    }
  }

  v8f c2 = {};
  #pragma unroll
  for (int kt = 0; kt < 2; ++kt) {
    v16h a2 = load_a_row(ldsH[wave], kt, 64, lane);
    c2 = wmma16(a2, load_bp(ldsW2, kt, lane), c2);
  }
  if (m < 7) {
    float bias = b2[m];
    #pragma unroll
    for (int j = 0; j < 8; ++j) {
      int row = j + 8 * g;
      int nd = nBase + row;
      if (nd < N) out[(size_t)nd * 7 + m] = c2[j] + bias;
    }
  }
}

static inline size_t align256(size_t v) { return (v + 255) & ~(size_t)255; }

extern "C" void kernel_launch(void* const* d_in, const int* in_sizes, int n_in,
                              void* d_out, int out_size, void* d_ws, size_t ws_size,
                              hipStream_t stream) {
  const float* x      = (const float*)d_in[0];
  const int*   ei     = (const int*)d_in[1];
  const float* ea     = (const float*)d_in[2];
  // d_in[3] = batch (unused)
  const float* hidden = (const float*)d_in[4];
  const float* e1_W1  = (const float*)d_in[5];
  const float* e1_b1  = (const float*)d_in[6];
  const float* e1_W2  = (const float*)d_in[7];
  const float* e1_b2  = (const float*)d_in[8];
  const float* rnnWih = (const float*)d_in[9];
  const float* rnnbih = (const float*)d_in[10];
  const float* rnnWhh = (const float*)d_in[11];
  const float* rnnbhh = (const float*)d_in[12];
  const float* e2_W1  = (const float*)d_in[13];
  const float* e2_b1  = (const float*)d_in[14];
  const float* e2_W2  = (const float*)d_in[15];
  const float* e2_b2  = (const float*)d_in[16];
  const float* n2_W1  = (const float*)d_in[17];
  const float* n2_b1  = (const float*)d_in[18];
  const float* n2_W2  = (const float*)d_in[19];
  const float* n2_b2  = (const float*)d_in[20];

  const int N = in_sizes[0] / 8;   // x is [N, 8]
  const int E = in_sizes[2] / 4;   // edge_attr is [E, 4]

  // Workspace layout
  char* ws = (char*)d_ws;
  size_t off = 0;
  float* agg1 = (float*)(ws + off); off = align256(off + (size_t)N * 32 * sizeof(float));
  float* agg2 = (float*)(ws + off); off = align256(off + (size_t)N * 32 * sizeof(float));
  h16*   x1c  = (h16*)(ws + off);   off = align256(off + (size_t)N * 72 * sizeof(h16));
  h16*   ef   = (h16*)(ws + off);   off = align256(off + (size_t)E * 48 * sizeof(h16));
  h16* pk1a = (h16*)(ws + off); off = align256(off + (size_t)PK1A * sizeof(h16));
  h16* pk1b = (h16*)(ws + off); off = align256(off + (size_t)PK1B * sizeof(h16));
  h16* pk2a = (h16*)(ws + off); off = align256(off + (size_t)PK2A * sizeof(h16));
  h16* pk2b = (h16*)(ws + off); off = align256(off + (size_t)PK2B * sizeof(h16));
  h16* pk3a = (h16*)(ws + off); off = align256(off + (size_t)PK3A * sizeof(h16));
  h16* pk3b = (h16*)(ws + off); off = align256(off + (size_t)PK3B * sizeof(h16));
  h16* pk4a = (h16*)(ws + off); off = align256(off + (size_t)PK4A * sizeof(h16));
  h16* pk4b = (h16*)(ws + off); off = align256(off + (size_t)PK4B * sizeof(h16));
  (void)ws_size;

  float* out  = (float*)d_out;                   // [N, 7]
  float* hOut = out + (size_t)N * 7;             // [N, 64]

  hipMemsetAsync(agg1, 0, (size_t)N * 32 * sizeof(float), stream);
  hipMemsetAsync(agg2, 0, (size_t)N * 32 * sizeof(float), stream);

  // Prepack all weight matrices into fragment-packed f16 (tiny, L2-resident)
  dim3 pblk(256);
  prepack_kernel<<<dim3((PK1A + 255) / 256), pblk, 0, stream>>>(e1_W1, pk1a, 64, 20, 1, 4);
  prepack_kernel<<<dim3((PK1B + 255) / 256), pblk, 0, stream>>>(e1_W2, pk1b, 32, 64, 2, 2);
  prepack_kernel<<<dim3((PK2A + 255) / 256), pblk, 0, stream>>>(rnnWih, pk2a, 192, 40, 2, 12);
  prepack_kernel<<<dim3((PK2B + 255) / 256), pblk, 0, stream>>>(rnnWhh, pk2b, 192, 64, 2, 12);
  prepack_kernel<<<dim3((PK3A + 255) / 256), pblk, 0, stream>>>(e2_W1, pk3a, 64, 180, 6, 4);
  prepack_kernel<<<dim3((PK3B + 255) / 256), pblk, 0, stream>>>(e2_W2, pk3b, 32, 64, 2, 2);
  prepack_kernel<<<dim3((PK4A + 255) / 256), pblk, 0, stream>>>(n2_W1, pk4a, 64, 104, 4, 4);
  prepack_kernel<<<dim3((PK4B + 255) / 256), pblk, 0, stream>>>(n2_W2, pk4b, 7, 64, 2, 1);

  const int WPB = 8;
  const int eTiles = (E + 15) / 16;
  const int nTiles = (N + 15) / 16;
  dim3 blk(256);
  dim3 gE((eTiles + WPB - 1) / WPB);
  dim3 gN((nTiles + WPB - 1) / WPB);

  gn1_edge_kernel<<<gE, blk, 0, stream>>>(x, ei, ea, pk1a, e1_b1, pk1b, e1_b2,
                                          ef, agg1, N, E);
  gru_node_kernel<<<gN, blk, 0, stream>>>(x, hidden, agg1, pk2a, rnnbih, pk2b, rnnbhh,
                                          hOut, x1c, N);
  gn2_edge_kernel<<<gE, blk, 0, stream>>>(ei, x1c, ef, pk3a, e2_b1, pk3b, e2_b2,
                                          agg2, N, E);
  out_node_kernel<<<gN, blk, 0, stream>>>(x1c, agg2, pk4a, n2_b1, pk4b, n2_b2, out, N);
}
